// MultiHeadAttention_472446403264
// MI455X (gfx1250) — compile-verified, matrix operand order checked by reading, not run on hardware
//
#include <hip/hip_runtime.h>
#include <hip/hip_fp16.h>

typedef _Float16 v16h __attribute__((ext_vector_type(16)));
typedef float    v8f  __attribute__((ext_vector_type(8)));
typedef int      v4i  __attribute__((ext_vector_type(4)));

#define C_DIM 1024
#define T_DIM 1024
#define H_DIM 16
#define KC    64
#define B_DIM 8

#if __has_builtin(__builtin_amdgcn_global_load_async_to_lds_b128) && \
    __has_builtin(__builtin_amdgcn_s_wait_asynccnt)
#define HAVE_ASYNC_LDS 1
typedef __attribute__((address_space(1))) v4i* gv4i_ptr;
typedef __attribute__((address_space(3))) v4i* lds_v4i_ptr;
#else
#define HAVE_ASYNC_LDS 0
#endif

union Frag16 {
  v16h     v;
  _Float16 h[16];
  unsigned u[8];
};

__device__ __forceinline__ void wait_lds() {
#if __has_builtin(__builtin_amdgcn_s_wait_dscnt)
  __builtin_amdgcn_s_wait_dscnt(0);
#else
  asm volatile("s_wait_dscnt 0x0" ::: "memory");
#endif
}

// pack two f32 into one dword of two f16 (lowers to v_cvt_pk_f16_f32)
__device__ __forceinline__ unsigned pack_h2(float a, float b) {
  union { _Float16 h[2]; unsigned u; } t;
  t.h[0] = (_Float16)a;
  t.h[1] = (_Float16)b;
  return t.u;
}

// -----------------------------------------------------------------------------
// GEMM: D[M=1024, N=1024] = W[M,K=1024] * X[K,N] + bias, per batch (grid z).
// Block 256 threads (8 waves), tile 64(M) x 128(N). f32 -> f16 conversion done
// ONCE at stage time: A staged as f16 rows (fragment = 2x ds_load_b128),
// B staged as packed K-pair dwords (fragment element = 1x ds_load_b32).
// mode 0: f16 out, head-major [B,H,T,kc], val=(acc+bias)*scale (Q-scale folded)
// mode 1: f32 out, [B,C,T] (final projection)
// mode 2: f16 out, pair-packed [B,H,T/2,kc,2] (V for packed PV fragments)
// -----------------------------------------------------------------------------
__global__ __launch_bounds__(256)
void gemm_wmma_kernel(const float* __restrict__ W,
                      const float* __restrict__ X,
                      const float* __restrict__ bias,
                      float scale, int mode,
                      _Float16* __restrict__ out_h,
                      float* __restrict__ out_f)
{
  constexpr int LDA = 40;   // f16 row stride for Ah[64][40] (80B, 16B-aligned)
  constexpr int LDB = 132;  // dword row stride for Bp[16][132] (528B, 16B-aligned)
  __shared__ __align__(16) _Float16 Ah[64 * LDA];
  __shared__ __align__(16) unsigned Bp[16 * LDB];

  const int b    = blockIdx.z;
  const int m0   = blockIdx.y * 64;
  const int n0   = blockIdx.x * 128;
  const int tid  = threadIdx.x;
  const int wave = tid >> 5;
  const int lane = tid & 31;
  const int hh   = lane >> 4;   // half-wave id (K-split of A/B fragments)
  const int r    = lane & 15;

  const float* Xb = X + (size_t)b * C_DIM * T_DIM;

  v8f acc[4] = {};

  for (int k0 = 0; k0 < C_DIM; k0 += 32) {
    // stage A: 64x32 of W -> f16 (coalesced float4 loads, b64 LDS stores)
#pragma unroll
    for (int i = 0; i < 2; ++i) {
      int q   = i * 256 + tid;          // 0..511
      int row = q >> 3;
      int c4  = (q & 7) << 2;
      float4 v = *(const float4*)(W + (size_t)(m0 + row) * C_DIM + k0 + c4);
      union { unsigned u[2]; float2 f; } st;
      st.u[0] = pack_h2(v.x, v.y);
      st.u[1] = pack_h2(v.z, v.w);
      *(float2*)&Ah[row * LDA + c4] = st.f;
    }
    // stage B: 32x128 of X -> packed K-pair dwords (b128 LDS stores)
#pragma unroll
    for (int i = 0; i < 2; ++i) {
      int q  = i * 256 + tid;           // 0..511
      int kp = q >> 5;                  // K-pair 0..15
      int c4 = (q & 31) << 2;
      const float* xr = Xb + (size_t)(k0 + 2 * kp) * T_DIM + n0 + c4;
      float4 r0 = *(const float4*)xr;
      float4 r1 = *(const float4*)(xr + T_DIM);
      union { unsigned u[4]; uint4 q4; } pk;
      pk.u[0] = pack_h2(r0.x, r1.x);
      pk.u[1] = pack_h2(r0.y, r1.y);
      pk.u[2] = pack_h2(r0.z, r1.z);
      pk.u[3] = pack_h2(r0.w, r1.w);
      *(uint4*)&Bp[kp * LDB + c4] = pk.q4;
    }
    __syncthreads();

    // B fragment: element e2 -> K-pair 4*hh + (e2&3) + 8*(e2>=4)
    Frag16 bfrag;
    const int bcol = wave * 16 + r;
#pragma unroll
    for (int e2 = 0; e2 < 8; ++e2) {
      int kp = 4 * hh + (e2 & 3) + ((e2 & 4) ? 8 : 0);
      bfrag.u[e2] = Bp[kp * LDB + bcol];
    }
#pragma unroll
    for (int mt = 0; mt < 4; ++mt) {
      Frag16 afrag;
      const _Float16* ap = &Ah[(mt * 16 + r) * LDA + 8 * hh];
      *(uint4*)&afrag.u[0] = *(const uint4*)ap;          // K = 8hh .. 8hh+7
      *(uint4*)&afrag.u[4] = *(const uint4*)(ap + 16);   // K = 16+8hh .. +7
      acc[mt] = __builtin_amdgcn_wmma_f32_16x16x32_f16(
          false, afrag.v, false, bfrag.v, (short)0, acc[mt], false, false);
    }
    __syncthreads();
  }

  // epilogue: C/D layout -> row m = f + 8*hh, col n = r
  const int ocol = wave * 16 + r;
#pragma unroll
  for (int mt = 0; mt < 4; ++mt) {
#pragma unroll
    for (int f = 0; f < 8; ++f) {
      int gm    = m0 + mt * 16 + f + 8 * hh;
      int gn    = n0 + ocol;
      float val = (acc[mt][f] + bias[gm]) * scale;
      if (mode == 0) {
        int head = gm >> 6, cc = gm & (KC - 1);
        out_h[(((size_t)b * H_DIM + head) * T_DIM + gn) * KC + cc] = (_Float16)val;
      } else if (mode == 1) {
        out_f[((size_t)b * C_DIM + gm) * T_DIM + gn] = val;
      } else {
        int head = gm >> 6, cc = gm & (KC - 1);
        size_t idx = (((size_t)b * H_DIM + head) * (T_DIM / 2) + (gn >> 1)) * (2 * KC)
                     + 2 * cc + (gn & 1);
        out_h[idx] = (_Float16)val;
      }
    }
  }
}

// -----------------------------------------------------------------------------
// Fused windowed-relative attention (flash style).
// Block = (b, h, 128-row query strip); 8 waves x 16 query rows.
// Q fragments persistent; K (row-major f16) and V (pair-packed f16) streamed
// through LDS in 32-key blocks, staged with CDNA5 async global->LDS copies.
// scores[i,j] = Qs[i].K[j] + Rq[i][clamp(j-i+4,0,8)]   (Qs pre-scaled 1/8)
// -----------------------------------------------------------------------------
__global__ __launch_bounds__(256)
void attn_rel_kernel(const _Float16* __restrict__ Q,
                     const _Float16* __restrict__ K,
                     const _Float16* __restrict__ V,
                     const float* __restrict__ emb_rel_k,
                     float* __restrict__ O)
{
  __shared__ __align__(16) _Float16 Ks[32][64];
  __shared__ __align__(16) _Float16 Vs[32 * 64];   // pair-packed: dword[16][64]
  __shared__ __align__(16) _Float16 Ps[8][16][32];
  __shared__ float Rq[8][16][9];

  const int b    = blockIdx.z;
  const int h    = blockIdx.y;
  const int q0   = blockIdx.x * 128;
  const int tid  = threadIdx.x;
  const int wave = tid >> 5;
  const int lane = tid & 31;
  const int hh   = lane >> 4;
  const int r    = lane & 15;

  const _Float16* Qbh = Q + (size_t)(b * H_DIM + h) * T_DIM * KC;
  const _Float16* Kbh = K + (size_t)(b * H_DIM + h) * T_DIM * KC;
  const _Float16* Vbh = V + (size_t)(b * H_DIM + h) * T_DIM * KC;  // packed, same extent

  // relative-bias table: Rq[wave][row][m] = Qs[row] . emb_rel_k[m], m in 0..8
  for (int e = lane; e < 16 * 9; e += 32) {
    int rr = e / 9, m = e % 9;
    const _Float16* qp = Qbh + (size_t)(q0 + wave * 16 + rr) * KC;
    const float*    ep = emb_rel_k + m * KC;
    float s = 0.f;
#pragma unroll 8
    for (int cix = 0; cix < KC; ++cix) s += (float)qp[cix] * ep[cix];
    Rq[wave][rr][m] = s;
  }

  // persistent Q A-fragments: 16 rows x 64 channels = 2 k-steps
  Frag16 qf[2];
  {
    const _Float16* qp = Qbh + (size_t)(q0 + wave * 16 + r) * KC;
#pragma unroll
    for (int ks = 0; ks < 2; ++ks) {
      *(uint4*)&qf[ks].u[0] = *(const uint4*)(qp + ks * 32 + 8 * hh);
      *(uint4*)&qf[ks].u[4] = *(const uint4*)(qp + ks * 32 + 8 * hh + 16);
    }
  }

  v8f oacc[4] = {};
  float mst[8], lst[8];
#pragma unroll
  for (int f = 0; f < 8; ++f) { mst[f] = -3.0e38f; lst[f] = 0.f; }

  for (int j0 = 0; j0 < T_DIM; j0 += 32) {
    // stage 32x64 f16 K and pair-packed V blocks (256 x 16B each)
#if HAVE_ASYNC_LDS
    __builtin_amdgcn_global_load_async_to_lds_b128(
        (gv4i_ptr)(Kbh + (size_t)j0 * KC + (size_t)tid * 8),
        (lds_v4i_ptr)(&Ks[0][0] + tid * 8), 0, 0);
    __builtin_amdgcn_global_load_async_to_lds_b128(
        (gv4i_ptr)(Vbh + (size_t)j0 * KC + (size_t)tid * 8),
        (lds_v4i_ptr)(&Vs[0] + tid * 8), 0, 0);
    __builtin_amdgcn_s_wait_asynccnt(0);
#else
    ((float4*)&Ks[0][0])[tid] = ((const float4*)(Kbh + (size_t)j0 * KC))[tid];
    ((float4*)&Vs[0])[tid]    = ((const float4*)(Vbh + (size_t)j0 * KC))[tid];
#endif
    if (wave == 0 && j0 + 32 < T_DIM) {
      __builtin_prefetch(Kbh + (size_t)(j0 + 32 + lane) * KC, 0, 1);
      __builtin_prefetch(Vbh + (size_t)(j0 + 32 + lane) * KC, 0, 1);
    }
    __syncthreads();

    // scores for two 16-key subtiles (kc=64 -> 2 k-steps each)
    v8f sacc[2];
#pragma unroll
    for (int sub = 0; sub < 2; ++sub) {
      Frag16 kf0, kf1;
      const _Float16* kp0 = &Ks[sub * 16 + r][8 * hh];
      *(uint4*)&kf0.u[0] = *(const uint4*)kp0;
      *(uint4*)&kf0.u[4] = *(const uint4*)(kp0 + 16);
      *(uint4*)&kf1.u[0] = *(const uint4*)(kp0 + 32);
      *(uint4*)&kf1.u[4] = *(const uint4*)(kp0 + 48);
      v8f s = {};
      s = __builtin_amdgcn_wmma_f32_16x16x32_f16(false, qf[0].v, false, kf0.v,
                                                 (short)0, s, false, false);
      s = __builtin_amdgcn_wmma_f32_16x16x32_f16(false, qf[1].v, false, kf1.v,
                                                 (short)0, s, false, false);
      sacc[sub] = s;
    }

    // add windowed relative bias
#pragma unroll
    for (int sub = 0; sub < 2; ++sub)
#pragma unroll
      for (int f = 0; f < 8; ++f) {
        int m  = f + 8 * hh;
        int qi = q0 + wave * 16 + m;
        int kj = j0 + sub * 16 + r;
        int d  = kj - qi + 4;
        d = d < 0 ? 0 : (d > 8 ? 8 : d);
        sacc[sub][f] += Rq[wave][m][d];
      }

    // online softmax over this 32-key block (reduce across each 16-lane half)
    float corr[8];
#pragma unroll
    for (int f = 0; f < 8; ++f) {
      float mx = fmaxf(sacc[0][f], sacc[1][f]);
#pragma unroll
      for (int off = 1; off < 16; off <<= 1) mx = fmaxf(mx, __shfl_xor(mx, off, 32));
      float mnew = fmaxf(mst[f], mx);
      float cf   = __expf(mst[f] - mnew);
      float p0   = __expf(sacc[0][f] - mnew);
      float p1   = __expf(sacc[1][f] - mnew);
      float rs   = p0 + p1;
#pragma unroll
      for (int off = 1; off < 16; off <<= 1) rs += __shfl_xor(rs, off, 32);
      lst[f]  = lst[f] * cf + rs;
      mst[f]  = mnew;
      corr[f] = cf;
      sacc[0][f] = p0;
      sacc[1][f] = p1;
    }
#pragma unroll
    for (int g = 0; g < 4; ++g)
#pragma unroll
      for (int f = 0; f < 8; ++f) oacc[g][f] *= corr[f];

    // transpose P (D layout -> A layout) through this wave's private LDS region
#pragma unroll
    for (int sub = 0; sub < 2; ++sub)
#pragma unroll
      for (int f = 0; f < 8; ++f)
        Ps[wave][f + 8 * hh][sub * 16 + r] = (_Float16)sacc[sub][f];
    wait_lds();

    Frag16 pf;
    {
      const _Float16* pp = &Ps[wave][r][8 * hh];
      *(uint4*)&pf.u[0] = *(const uint4*)pp;
      *(uint4*)&pf.u[4] = *(const uint4*)(pp + 16);
    }

    // O[16 x 64] += P[16 x 32] * V[32 x 64]; V fragments from packed dwords
    const unsigned* Vp = (const unsigned*)&Vs[0];    // [16 kpairs][64 cc]
#pragma unroll
    for (int g = 0; g < 4; ++g) {
      Frag16 vf;
      int cc = g * 16 + r;
#pragma unroll
      for (int e2 = 0; e2 < 8; ++e2) {
        int kp = 4 * hh + (e2 & 3) + ((e2 & 4) ? 8 : 0);
        vf.u[e2] = Vp[kp * 64 + cc];
      }
      oacc[g] = __builtin_amdgcn_wmma_f32_16x16x32_f16(
          false, pf.v, false, vf.v, (short)0, oacc[g], false, false);
    }
    __syncthreads();
  }

  // normalize and store O as f32 [B, C, T] for the final projection GEMM
#pragma unroll
  for (int g = 0; g < 4; ++g)
#pragma unroll
    for (int f = 0; f < 8; ++f) {
      int m  = f + 8 * hh;
      int qi = q0 + wave * 16 + m;
      int cc = g * 16 + r;
      O[((size_t)b * C_DIM + h * KC + cc) * T_DIM + qi] = oacc[g][f] / lst[f];
    }
}

// -----------------------------------------------------------------------------
extern "C" void kernel_launch(void* const* d_in, const int* in_sizes, int n_in,
                              void* d_out, int out_size, void* d_ws, size_t ws_size,
                              hipStream_t stream)
{
  (void)in_sizes; (void)n_in; (void)out_size; (void)ws_size;

  const float* x     = (const float*)d_in[0];
  const float* c     = (const float*)d_in[1];
  const float* wq    = (const float*)d_in[2];
  const float* bq    = (const float*)d_in[3];
  const float* wk    = (const float*)d_in[4];
  const float* bk    = (const float*)d_in[5];
  const float* wv    = (const float*)d_in[6];
  const float* bv    = (const float*)d_in[7];
  const float* wo    = (const float*)d_in[8];
  const float* bo    = (const float*)d_in[9];
  const float* emb_k = (const float*)d_in[10];
  // d_in[11] (emb_rel_v) unused by the reference

  const size_t nqkv = (size_t)B_DIM * H_DIM * T_DIM * KC;  // 8Mi elements
  _Float16* Qh  = (_Float16*)d_ws;
  _Float16* Kh  = Qh + nqkv;
  _Float16* Vh  = Kh + nqkv;
  float*    Ows = (float*)(Vh + nqkv);
  float*    out = (float*)d_out;

  dim3 blk(256);
  dim3 ggrid(T_DIM / 128, C_DIM / 64, B_DIM);
  const float qscale = 0.125f;  // 1/sqrt(kc=64)

  gemm_wmma_kernel<<<ggrid, blk, 0, stream>>>(wq, x, bq, qscale, 0, Qh, nullptr);
  gemm_wmma_kernel<<<ggrid, blk, 0, stream>>>(wk, c, bk, 1.0f,   0, Kh, nullptr);
  gemm_wmma_kernel<<<ggrid, blk, 0, stream>>>(wv, c, bv, 1.0f,   2, Vh, nullptr);

  dim3 agrid(T_DIM / 128, H_DIM, B_DIM);
  attn_rel_kernel<<<agrid, blk, 0, stream>>>(Qh, Kh, Vh, emb_k, Ows);

  gemm_wmma_kernel<<<ggrid, blk, 0, stream>>>(wo, Ows, bo, 1.0f, 1, nullptr, out);
}
